// YunaAttention_90598040141885
// MI455X (gfx1250) — matrix operand order repaired and checked by reading, not compiled, not run on hardware
//
#include <hip/hip_runtime.h>
#include <hip/hip_bf16.h>
#include <stdint.h>

typedef __attribute__((ext_vector_type(16))) _Float16 v16h;
typedef __attribute__((ext_vector_type(8)))  float    v8f;
typedef __attribute__((ext_vector_type(4)))  int      i32x4;

#define T_SEQ 2048
#define C_EMB 2048
#define HEADS 16
#define KVH   4
#define HD    128
#define KVDIM 512   // KVH*HD

union FragH { v16h h; uint32_t u[8]; };

__device__ __forceinline__ v8f zero_v8f() {
  v8f z;
#pragma unroll
  for (int i = 0; i < 8; ++i) z[i] = 0.0f;
  return z;
}

// ---------------- CDNA5 async global->LDS copy path ----------------
#if defined(__AMDGCN__) && __has_builtin(__builtin_amdgcn_global_load_async_to_lds_b128)
#define HAVE_ASYNC_LDS 1
#else
#define HAVE_ASYNC_LDS 0
#endif

// copy 16B from global to LDS (async when available)
__device__ __forceinline__ void cp_b128(const _Float16* g, _Float16* l) {
#if HAVE_ASYNC_LDS
  typedef __attribute__((address_space(1))) i32x4 GVec;  // global int32x4
  typedef __attribute__((address_space(3))) i32x4 LVec;  // LDS int32x4
  GVec* gp = (GVec*)(__attribute__((address_space(1))) void*)(void*)g;
  LVec* lp = (LVec*)(__attribute__((address_space(3))) void*)(void*)l;
  __builtin_amdgcn_global_load_async_to_lds_b128(gp, lp, 0, 0);
#else
  *(uint4*)l = *(const uint4*)g;
#endif
}

// copy 32B (one 16-wide f16 segment) global -> LDS
__device__ __forceinline__ void cp_b256(const _Float16* g, _Float16* l) {
  cp_b128(g, l);
  cp_b128(g + 8, l + 8);
}

template <int N>
__device__ __forceinline__ void async_wait() {
#if defined(__AMDGCN__)
#if __has_builtin(__builtin_amdgcn_s_wait_asynccnt)
  __builtin_amdgcn_s_wait_asynccnt(N);
#else
  asm volatile("s_wait_asynccnt %0" ::"i"(N) : "memory");
#endif
#endif
}

// ---------------- WMMA fragment builders ----------------
// A fragment (16x32 f16): lane holds row M=lane%16; half=lane/16;
// VGPR j<4 -> K=half*8+2j ; j>=4 -> K=16+half*8+2(j-4)
// rowp points at this lane's 32 contiguous f16 (K-major).
__device__ __forceinline__ v16h frag_a32(const _Float16* rowp, int hl) {
  FragH f;
  const uint32_t* p = (const uint32_t*)rowp;  // dword i = K pair (2i,2i+1)
  int b1 = hl * 4;
  int b2 = 8 + hl * 4;
#pragma unroll
  for (int j = 0; j < 4; ++j) f.u[j] = p[b1 + j];
#pragma unroll
  for (int j = 0; j < 4; ++j) f.u[4 + j] = p[b2 + j];
  return f.h;
}

// B fragment (32x16 f16): lane holds col N=lane%16; half=lane/16;
// VGPR j<4 -> K=half*8+2j ; j>=4 -> K=16+half*8+2(j-4)
__device__ __forceinline__ v16h frag_b32(const _Float16* rowp, int hl) {
  FragH f;
  const uint32_t* p = (const uint32_t*)rowp;
  int b1 = hl * 4;
  int b2 = 8 + hl * 4;
#pragma unroll
  for (int j = 0; j < 4; ++j) f.u[j] = p[b1 + j];
#pragma unroll
  for (int j = 0; j < 4; ++j) f.u[4 + j] = p[b2 + j];
  return f.h;
}

__device__ __forceinline__ v8f wmma_f16(v16h a, v16h b, v8f c) {
  return __builtin_amdgcn_wmma_f32_16x16x32_f16(false, a, false, b, (short)0, c, false, false);
}

// ---------------- converts ----------------
__global__ void __launch_bounds__(256) cvt_f16_kernel(const float* __restrict__ src,
                                                      _Float16* __restrict__ dst, int n) {
  int i = blockIdx.x * 256 + threadIdx.x;
  if (i < n) dst[i] = (_Float16)src[i];
}

// W [K][N] f32 -> Wt [N][K] f16 (32x32 tiles via LDS)
__global__ void __launch_bounds__(256) transpose_cvt_kernel(const float* __restrict__ W,
                                                            _Float16* __restrict__ Wt,
                                                            int K, int N) {
  __shared__ float tile[32][33];
  int nb = blockIdx.x * 32, kb = blockIdx.y * 32;
  int tx = threadIdx.x & 31, ty = threadIdx.x >> 5;
#pragma unroll
  for (int r = ty; r < 32; r += 8) tile[r][tx] = W[(size_t)(kb + r) * N + nb + tx];
  __syncthreads();
#pragma unroll
  for (int r = ty; r < 32; r += 8)
    Wt[(size_t)(nb + r) * K + kb + tx] = (_Float16)tile[tx][r];
}

// ---------------- WMMA GEMM: C[M,N] f32 = A[M,K]f16 @ Bt[N,K]f16 (+bias) ----------------
// 128x128 block, 8 waves (2x4), each wave 64x32 = 4x2 WMMA accumulators.
// K-tiles of 32 double-buffered in LDS via async global->LDS copies.
__global__ void __launch_bounds__(256) gemm_f16_kernel(const _Float16* __restrict__ A,
                                                       const _Float16* __restrict__ Bt,
                                                       const float* __restrict__ bias,
                                                       float* __restrict__ C,
                                                       int M, int N, int K) {
  __shared__ _Float16 As[2][128 * 32];
  __shared__ _Float16 Bs[2][128 * 32];
  int tid = threadIdx.x;
  int lane = tid & 31, wave = tid >> 5;
  int hl = lane >> 4, nlo = lane & 15;
  int wm = wave >> 2, wn = wave & 3;               // 2x4 wave grid
  int bm = blockIdx.x * 128, bn = blockIdx.y * 128;
  int lr = tid >> 1;
  int ls = (tid & 1) * 16;
  int sidx = lr * 32 + ls;
  const _Float16* Ag = A + (size_t)(bm + lr) * K + ls;
  const _Float16* Bg = Bt + (size_t)(bn + lr) * K + ls;
  v8f acc[4][2];
#pragma unroll
  for (int mi = 0; mi < 4; ++mi)
#pragma unroll
    for (int ni = 0; ni < 2; ++ni) acc[mi][ni] = zero_v8f();

  // prologue: stage tile 0
  cp_b256(Ag, &As[0][sidx]);
  cp_b256(Bg, &Bs[0][sidx]);

  int nk = K >> 5;
  for (int kt = 0; kt < nk; ++kt) {
    Ag += 32; Bg += 32;
    if (kt + 1 < nk) {           // stage next tile while current computes
      int nb = (kt + 1) & 1;
      cp_b256(Ag, &As[nb][sidx]);
      cp_b256(Bg, &Bs[nb][sidx]);
      async_wait<4>();           // current tile (4 older async ops) resident
    } else {
      async_wait<0>();
    }
    __syncthreads();
    const _Float16* Ab = As[kt & 1];
    const _Float16* Bb = Bs[kt & 1];
    v16h af[4], bf[2];
#pragma unroll
    for (int mi = 0; mi < 4; ++mi)
      af[mi] = frag_a32(&Ab[(wm * 64 + mi * 16 + nlo) * 32], hl);
#pragma unroll
    for (int ni = 0; ni < 2; ++ni)
      bf[ni] = frag_b32(&Bb[(wn * 32 + ni * 16 + nlo) * 32], hl);
#pragma unroll
    for (int mi = 0; mi < 4; ++mi)
#pragma unroll
      for (int ni = 0; ni < 2; ++ni)
        acc[mi][ni] = wmma_f16(af[mi], bf[ni], acc[mi][ni]);
    __syncthreads();             // all waves done with buf kt&1 before it is refilled
  }
#pragma unroll
  for (int mi = 0; mi < 4; ++mi)
#pragma unroll
    for (int ni = 0; ni < 2; ++ni)
#pragma unroll
      for (int i = 0; i < 8; ++i) {
        int row = bm + wm * 64 + mi * 16 + i + 8 * hl;
        int col = bn + wn * 32 + ni * 16 + nlo;
        float v = acc[mi][ni][i];
        if (bias) v += bias[col];
        C[(size_t)row * N + col] = v;
      }
}

// ---------------- RoPE ----------------
__device__ __forceinline__ int rope_chan(int d) {
  return (d < 16) ? 0 : (d < 40) ? 1 : (d < 64) ? 2 : (d < 80) ? 0 : (d < 104) ? 1 : 2;
}

// Q: [T, C] f32 -> rope (+1/sqrt(D) folded in) -> Qh f16 [H][T][D]
__global__ void __launch_bounds__(256) rope_q_kernel(const float* __restrict__ Qlin,
                                                     const float* __restrict__ cosp,
                                                     const float* __restrict__ sinp,
                                                     _Float16* __restrict__ Qh) {
  int idx = blockIdx.x * 256 + threadIdx.x;     // T*C
  int t = idx >> 11, col = idx & 2047;
  int h = col >> 7, d = col & 127;
  int ch = rope_chan(d);
  float c = cosp[(size_t)(ch * T_SEQ + t) * HD + d];
  float s = sinp[(size_t)(ch * T_SEQ + t) * HD + d];
  float q = Qlin[idx];
  int d2 = (d < 64) ? d + 64 : d - 64;
  float q2 = Qlin[(size_t)t * C_EMB + (col & ~127) + d2];
  float rot = (d < 64) ? -q2 : q2;
  const float scale = 0.08838834764831845f;     // 1/sqrt(128)
  Qh[((size_t)h * T_SEQ + t) * HD + d] = (_Float16)((q * c + rot * s) * scale);
}

// K,V: [T,512] f32 -> present_k/present_v f32 [Hkv][T][D] (into d_out) +
//      Kh f16 [Hkv][T][D], Vt f16 [Hkv][D][T]
__global__ void __launch_bounds__(256) rope_kv_kernel(const float* __restrict__ Klin,
                                                      const float* __restrict__ Vlin,
                                                      const float* __restrict__ cosp,
                                                      const float* __restrict__ sinp,
                                                      float* __restrict__ pk,
                                                      float* __restrict__ pv,
                                                      _Float16* __restrict__ Kh,
                                                      _Float16* __restrict__ Vt) {
  int idx = blockIdx.x * 256 + threadIdx.x;     // T*512
  int t = idx >> 9, col = idx & 511;
  int hk = col >> 7, d = col & 127;
  int ch = rope_chan(d);
  float c = cosp[(size_t)(ch * T_SEQ + t) * HD + d];
  float s = sinp[(size_t)(ch * T_SEQ + t) * HD + d];
  float k = Klin[idx];
  int d2 = (d < 64) ? d + 64 : d - 64;
  float k2 = Klin[(size_t)t * KVDIM + (col & ~127) + d2];
  float rot = (d < 64) ? -k2 : k2;
  float kr = k * c + rot * s;
  size_t hd_idx = ((size_t)hk * T_SEQ + t) * HD + d;
  pk[hd_idx] = kr;
  Kh[hd_idx] = (_Float16)kr;
  float v = Vlin[idx];
  pv[hd_idx] = v;
  Vt[((size_t)hk * HD + d) * T_SEQ + t] = (_Float16)v;
}

// ---------------- causal flash attention (GQA) ----------------
// grid: (qtiles=16, heads=16), block 256 (8 waves x 16 query rows)
// K/V 32-key tiles double-buffered in LDS via async copies.
__global__ void __launch_bounds__(256) attn_kernel(const _Float16* __restrict__ Qh,  // [H][T][D]
                                                   const _Float16* __restrict__ Kh,  // [Hkv][T][D]
                                                   const _Float16* __restrict__ Vt,  // [Hkv][D][T]
                                                   _Float16* __restrict__ Yh) {      // [T][H*D]
  __shared__ _Float16 Ks[2][32 * HD];    // [key][d]
  __shared__ _Float16 Vs[2][HD * 32];    // [d][key]
  __shared__ float    Ss[128 * 32];      // [row][key] scores
  __shared__ _Float16 Ps[128 * 32];      // [row][key] probs
  __shared__ float mrow[128], lrow[128], frow[128];

  int tid = threadIdx.x;
  int lane = tid & 31, w = tid >> 5;
  int hl = lane >> 4, nlo = lane & 15;
  int qt = blockIdx.x, h = blockIdx.y;
  int hkv = h >> 2;

  if (tid < 128) { mrow[tid] = -3.0e38f; lrow[tid] = 0.0f; }

  // preload this wave's 16 query rows as 4 A-fragments (d-chunks of 32)
  int qrow = qt * 128 + w * 16 + nlo;
  const _Float16* qp = Qh + ((size_t)h * T_SEQ + qrow) * HD;
  v16h qfrag[4];
#pragma unroll
  for (int dk = 0; dk < 4; ++dk) qfrag[dk] = frag_a32(qp + dk * 32, hl);

  v8f acc[8];
#pragma unroll
  for (int dn = 0; dn < 8; ++dn) acc[dn] = zero_v8f();

  // cooperative K/V staging positions for this thread
  int kr_ = tid >> 3, kseg = (tid & 7) * 16;
  int vd_ = tid >> 1, vseg = (tid & 1) * 16;
  const _Float16* Kbase = Kh + (size_t)hkv * T_SEQ * HD;
  const _Float16* Vbase = Vt + (size_t)hkv * HD * T_SEQ;

  // prologue: stage key-block 0
  cp_b256(Kbase + (size_t)kr_ * HD + kseg, &Ks[0][kr_ * HD + kseg]);
  cp_b256(Vbase + (size_t)vd_ * T_SEQ + vseg, &Vs[0][vd_ * 32 + vseg]);

  int nkb = qt * 4 + 4;   // key blocks of 32 covering keys <= qt*128+127
  for (int kb = 0; kb < nkb; ++kb) {
    if (kb + 1 < nkb) {
      int nb = (kb + 1) & 1;
      cp_b256(Kbase + (size_t)((kb + 1) * 32 + kr_) * HD + kseg, &Ks[nb][kr_ * HD + kseg]);
      cp_b256(Vbase + (size_t)vd_ * T_SEQ + (kb + 1) * 32 + vseg, &Vs[nb][vd_ * 32 + vseg]);
      async_wait<4>();
    } else {
      async_wait<0>();
    }
    __syncthreads();
    const _Float16* Kb = Ks[kb & 1];
    const _Float16* Vb = Vs[kb & 1];

    // S = Q @ K^T  (16 q-rows x 32 keys per wave)
    v8f sacc[2];
    sacc[0] = zero_v8f(); sacc[1] = zero_v8f();
#pragma unroll
    for (int dk = 0; dk < 4; ++dk) {
#pragma unroll
      for (int nk = 0; nk < 2; ++nk) {
        v16h bf = frag_b32(&Kb[(nk * 16 + nlo) * HD + dk * 32], hl);
        sacc[nk] = wmma_f16(qfrag[dk], bf, sacc[nk]);
      }
    }
    // spill with causal mask
#pragma unroll
    for (int nk = 0; nk < 2; ++nk)
#pragma unroll
      for (int i = 0; i < 8; ++i) {
        int r = i + 8 * hl;
        int colk = nk * 16 + nlo;
        int kg = kb * 32 + colk;
        int qg = qt * 128 + w * 16 + r;
        float sv = sacc[nk][i];
        if (kg > qg) sv = -1.0e30f;
        Ss[(w * 16 + r) * 32 + colk] = sv;
      }

    // per-row online softmax (lanes 0..15, each owns one row of this wave)
    if (lane < 16) {
      int ar = w * 16 + lane;
      float mo = mrow[ar];
      float mx = -3.0e38f;
#pragma unroll
      for (int c = 0; c < 32; ++c) mx = fmaxf(mx, Ss[ar * 32 + c]);
      float mn = fmaxf(mo, mx);
      float ef = __expf(mo - mn);
      float sum = 0.0f;
#pragma unroll
      for (int c = 0; c < 32; ++c) {
        float p = __expf(Ss[ar * 32 + c] - mn);
        sum += p;
        Ps[ar * 32 + c] = (_Float16)p;
      }
      mrow[ar] = mn;
      lrow[ar] = lrow[ar] * ef + sum;
      frow[ar] = ef;
    }

    // rescale running accumulators and add P @ V
    float fr[8];
#pragma unroll
    for (int i = 0; i < 8; ++i) fr[i] = frow[w * 16 + i + 8 * hl];
    v16h pa = frag_a32(&Ps[(w * 16 + nlo) * 32], hl);
#pragma unroll
    for (int dn = 0; dn < 8; ++dn) {
#pragma unroll
      for (int i = 0; i < 8; ++i) acc[dn][i] *= fr[i];
      v16h vb = frag_b32(&Vb[(dn * 16 + nlo) * 32], hl);
      acc[dn] = wmma_f16(pa, vb, acc[dn]);
    }
    __syncthreads();   // all waves done with buf kb&1 before refill
  }

  // normalize and write y (f16, [T][H*D] for the final WMMA GEMM)
  float li[8];
#pragma unroll
  for (int i = 0; i < 8; ++i) li[i] = lrow[w * 16 + i + 8 * hl];
#pragma unroll
  for (int dn = 0; dn < 8; ++dn)
#pragma unroll
    for (int i = 0; i < 8; ++i) {
      int qg = qt * 128 + w * 16 + i + 8 * hl;
      int d = dn * 16 + nlo;
      Yh[(size_t)qg * C_EMB + h * HD + d] = (_Float16)(acc[dn][i] / li[i]);
    }
}

// ---------------- host launch ----------------
extern "C" void kernel_launch(void* const* d_in, const int* in_sizes, int n_in,
                              void* d_out, int out_size, void* d_ws, size_t ws_size,
                              hipStream_t stream) {
  (void)in_sizes; (void)n_in; (void)out_size; (void)ws_size;
  const float* x    = (const float*)d_in[0];
  const float* cosp = (const float*)d_in[1];
  const float* sinp = (const float*)d_in[2];
  const float* Wq   = (const float*)d_in[3];
  const float* bq   = (const float*)d_in[4];
  const float* Wk   = (const float*)d_in[5];
  const float* bk   = (const float*)d_in[6];
  const float* Wv   = (const float*)d_in[7];
  const float* bv   = (const float*)d_in[8];
  const float* Wo   = (const float*)d_in[9];

  float* y_out  = (float*)d_out;
  float* pk_out = y_out + (size_t)T_SEQ * C_EMB;
  float* pv_out = pk_out + (size_t)KVH * T_SEQ * HD;

  char* ws = (char*)d_ws;
  size_t off = 0;
  auto alloc = [&](size_t bytes) -> char* {
    char* p = ws + off;
    off += (bytes + 255) & ~(size_t)255;
    return p;
  };
  _Float16* xh   = (_Float16*)alloc((size_t)T_SEQ * C_EMB * 2);
  _Float16* Wqt  = (_Float16*)alloc((size_t)C_EMB * C_EMB * 2);
  _Float16* Wkt  = (_Float16*)alloc((size_t)KVDIM * C_EMB * 2);
  _Float16* Wvt  = (_Float16*)alloc((size_t)KVDIM * C_EMB * 2);
  _Float16* Wot  = (_Float16*)alloc((size_t)C_EMB * C_EMB * 2);
  float*    Qlin = (float*)alloc((size_t)T_SEQ * C_EMB * 4);
  float*    Klin = (float*)alloc((size_t)T_SEQ * KVDIM * 4);
  float*    Vlin = (float*)alloc((size_t)T_SEQ * KVDIM * 4);
  _Float16* Qh   = (_Float16*)alloc((size_t)HEADS * T_SEQ * HD * 2);
  _Float16* Kh   = (_Float16*)alloc((size_t)KVH * T_SEQ * HD * 2);
  _Float16* Vt   = (_Float16*)alloc((size_t)KVH * HD * T_SEQ * 2);
  _Float16* Yh   = (_Float16*)alloc((size_t)T_SEQ * C_EMB * 2);

  // converts
  cvt_f16_kernel<<<(T_SEQ * C_EMB + 255) / 256, 256, 0, stream>>>(x, xh, T_SEQ * C_EMB);
  transpose_cvt_kernel<<<dim3(C_EMB / 32, C_EMB / 32), 256, 0, stream>>>(Wq, Wqt, C_EMB, C_EMB);
  transpose_cvt_kernel<<<dim3(KVDIM / 32, C_EMB / 32), 256, 0, stream>>>(Wk, Wkt, C_EMB, KVDIM);
  transpose_cvt_kernel<<<dim3(KVDIM / 32, C_EMB / 32), 256, 0, stream>>>(Wv, Wvt, C_EMB, KVDIM);
  transpose_cvt_kernel<<<dim3(C_EMB / 32, C_EMB / 32), 256, 0, stream>>>(Wo, Wot, C_EMB, C_EMB);

  // projections
  gemm_f16_kernel<<<dim3(T_SEQ / 128, C_EMB / 128), 256, 0, stream>>>(xh, Wqt, bq, Qlin, T_SEQ, C_EMB, C_EMB);
  gemm_f16_kernel<<<dim3(T_SEQ / 128, KVDIM / 128), 256, 0, stream>>>(xh, Wkt, bk, Klin, T_SEQ, KVDIM, C_EMB);
  gemm_f16_kernel<<<dim3(T_SEQ / 128, KVDIM / 128), 256, 0, stream>>>(xh, Wvt, bv, Vlin, T_SEQ, KVDIM, C_EMB);

  // rope + layout (also fills present_k / present_v regions of d_out)
  rope_q_kernel<<<(T_SEQ * C_EMB) / 256, 256, 0, stream>>>(Qlin, cosp, sinp, Qh);
  rope_kv_kernel<<<(T_SEQ * KVDIM) / 256, 256, 0, stream>>>(Klin, Vlin, cosp, sinp, pk_out, pv_out, Kh, Vt);

  // flash attention
  attn_kernel<<<dim3(T_SEQ / 128, HEADS), 256, 0, stream>>>(Qh, Kh, Vt, Yh);

  // output projection
  gemm_f16_kernel<<<dim3(T_SEQ / 128, C_EMB / 128), 256, 0, stream>>>(Yh, Wot, nullptr, y_out, T_SEQ, C_EMB, C_EMB);
}
